// model2_27814208209093
// MI455X (gfx1250) — compile-verified, matrix operand order checked by reading, not run on hardware
//
#include <hip/hip_runtime.h>

typedef __attribute__((ext_vector_type(16))) _Float16 v16h;
typedef __attribute__((ext_vector_type(8)))  float    v8f;
typedef __attribute__((ext_vector_type(4)))  float    v4f;
typedef __attribute__((ext_vector_type(4)))  unsigned int u32x4;
typedef __attribute__((ext_vector_type(8)))  int      i32x8;
typedef __attribute__((ext_vector_type(4)))  int      i32x4;

#define Hh 16
#define Bn 1024
#define Tn 512
#define Dn 64
#define BT 16        // sequences per workgroup
#define NTHREADS 512 // 16 waves (wave32)

static __device__ __forceinline__ v8f wmma_f16(v16h a, v16h b, v8f c) {
  // D = A(16x32 f16) * B(32x16 f16) + C(16x16 f32)
  return __builtin_amdgcn_wmma_f32_16x16x32_f16(false, a, false, b, (short)0, c,
                                                false, false);
}

// TDM: DMA one [16 x 64] f32 tile (rows strided by T*D elements) into LDS.
// D# group1 constants: data_size=4B, tensor_dim0=64, tensor_dim1=16,
// tile_dim0=64, tile_dim1=16, tensor_dim0_stride=T*D=32768.
static __device__ __forceinline__ void tdm_load_tile(const float* gptr,
                                                     unsigned int lds_off) {
  unsigned long long ga = (unsigned long long)(uintptr_t)gptr;
  u32x4 g0;
  g0[0] = 1u;                                  // count=1 (valid descriptor)
  g0[1] = lds_off;                             // lds_addr (bytes)
  g0[2] = (unsigned int)ga;                    // global_addr[31:0]
  g0[3] = (unsigned int)((ga >> 32) & 0x01FFFFFFu) | 0x80000000u; // [56:32]|type=2
  i32x8 g1 = {0x00020000, 0x00400000, 0x00100000, 0x00400000,
              0x00000010, 0x00008000, 0, 0};
  i32x4 z4 = {0, 0, 0, 0};
  i32x8 z8 = {0, 0, 0, 0, 0, 0, 0, 0};
  __builtin_amdgcn_tensor_load_to_lds(g0, g1, z4, z4, z8, 0);
}

// Build the emission A-matrix (16x32 f16 per K-chunk) from the raw f32 tile.
static __device__ __forceinline__ void build_yA(const float* yb, int hs,
                                                v16h& Y0, v16h& Y1) {
  v4f g0a = *(const v4f*)(yb + 8 * hs);
  v4f g0b = *(const v4f*)(yb + 8 * hs + 4);
  v4f g1a = *(const v4f*)(yb + 16 + 8 * hs);
  v4f g1b = *(const v4f*)(yb + 16 + 8 * hs + 4);
  v4f g2a = *(const v4f*)(yb + 32 + 8 * hs);
  v4f g2b = *(const v4f*)(yb + 32 + 8 * hs + 4);
  v4f g3a = *(const v4f*)(yb + 48 + 8 * hs);
  v4f g3b = *(const v4f*)(yb + 48 + 8 * hs + 4);
#pragma unroll
  for (int e = 0; e < 4; ++e) {
    Y0[e]      = (_Float16)g0a[e];
    Y0[e + 4]  = (_Float16)g0b[e];
    Y0[e + 8]  = (_Float16)g1a[e];
    Y0[e + 12] = (_Float16)g1b[e];
    Y1[e]      = (_Float16)g2a[e];
    Y1[e + 4]  = (_Float16)g2b[e];
    Y1[e + 8]  = (_Float16)g3a[e];
    Y1[e + 12] = (_Float16)g3b[e];
  }
}

__global__ __launch_bounds__(NTHREADS) void fhmm_forward(
    const float* __restrict__ seq,      // [B, T, D]
    const int*   __restrict__ lengths,  // [B]
    const float* __restrict__ pw,       // [H, H]
    const float* __restrict__ px,       // [H, H]
    const float* __restrict__ py,       // [H, H, D]
    float*       __restrict__ out)      // [B]
{
  __shared__ float alpha_s[BT * Hh * Hh];             // [b][w][x]  16 KB
  __shared__ float aw_s[BT * Hh * Hh];                // [b][w'][x] 16 KB
  __shared__ __align__(16) float yraw[2][BT * Dn];    // f32 y tiles, 8 KB

  const int tid  = threadIdx.x;
  const int j    = tid >> 5;   // wave id = x-slice (phase A) / w'-slice (phase B)
  const int lane = tid & 31;
  const int lp   = lane & 15;
  const int hs   = lane >> 4;  // half-wave select
  const int bg0  = blockIdx.x * BT;
  const float* seq_wg = seq + ((size_t)bg0) * Tn * Dn;  // tiles at seq_wg + t*Dn

  // ---------------- constant matrices into registers ----------------
  // Transition B-matrices, hi/lo f16 split packed into K (A carries ea_hi in
  // K<16 and ea_lo in K>=16; second WMMA with the lo matrix restores f32 acc).
  v16h pwB1, pwB2, pxB1, pxB2;
#pragma unroll
  for (int r = 0; r < 8; ++r) {
    const int w = 8 * hs + r;
    float v = pw[w * Hh + lp];
    _Float16 h = (_Float16)v, l = (_Float16)(v - (float)h);
    pwB1[r] = h; pwB1[r + 8] = h; pwB2[r] = l; pwB2[r + 8] = l;
    v = px[w * Hh + lp];
    h = (_Float16)v; l = (_Float16)(v - (float)h);
    pxB1[r] = h; pxB1[r + 8] = h; pxB2[r] = l; pxB2[r + 8] = l;
  }

  // Emission B-matrices for this wave's w = j slice: diff[j][x'=lp][d], hi/lo.
  v16h dBh0, dBh1, dBl0, dBl1;
  float basep = 0.f;
#pragma unroll
  for (int e = 0; e < 16; ++e) {
    const int dd = 8 * hs + (e & 7) + ((e & 8) ? 16 : 0);
    float p0 = py[(j * Hh + lp) * Dn + dd];
    float l1m0 = log1pf(-p0);
    float df0 = logf(p0) - l1m0;
    _Float16 h0 = (_Float16)df0;
    dBh0[e] = h0; dBl0[e] = (_Float16)(df0 - (float)h0);
    float p1 = py[(j * Hh + lp) * Dn + 32 + dd];
    float l1m1 = log1pf(-p1);
    float df1 = logf(p1) - l1m1;
    _Float16 h1 = (_Float16)df1;
    dBh1[e] = h1; dBl1[e] = (_Float16)(df1 - (float)h1);
    basep += l1m0 + l1m1;
  }
  const float base_j = basep + __shfl_xor(basep, 16, 32);

  int lenv[8];
#pragma unroll
  for (int r = 0; r < 8; ++r) lenv[r] = lengths[bg0 + r + 8 * hs];

  // ---------------- DMA y tile t=0 into LDS (wave 0) ----------------
  if (j == 0) {
    tdm_load_tile(seq_wg + 0 * Dn, (unsigned int)(uintptr_t)&yraw[0][0]);
    __builtin_amdgcn_s_wait_tensorcnt(0);
  }
  __syncthreads();

  // ---------------- t = 0 init: alpha0 = lpw0[w] + lpx0[x] + emit(y0) -------
  {
    if (j == 0)  // overlap: DMA tile t=1 while computing init
      tdm_load_tile(seq_wg + 1 * Dn, (unsigned int)(uintptr_t)&yraw[1][0]);

    v16h Y0, Y1;
    build_yA(&yraw[0][lp * Dn], hs, Y0, Y1);
    v8f E = {};
    E = wmma_f16(Y0, dBh0, E);
    E = wmma_f16(Y1, dBh1, E);
    E = wmma_f16(Y0, dBl0, E);
    E = wmma_f16(Y1, dBl1, E);
    const float lw0 = logf(pw[j]);   // log pw[0][j]
    const float lx0 = logf(px[lp]);  // log px[0][lp]
#pragma unroll
    for (int r = 0; r < 8; ++r) {
      const int b = r + 8 * hs;
      alpha_s[b * 256 + j * 16 + lp] = lw0 + lx0 + base_j + E[r];
    }
    if (j == 0) __builtin_amdgcn_s_wait_tensorcnt(0);
  }
  __syncthreads();

  // ---------------- main recurrence ----------------
  for (int t = 1; t < Tn; ++t) {
    // async DMA of tile t+1 into the other buffer; overlaps phases A/B
    if ((t + 1 < Tn) && (j == 0))
      tdm_load_tile(seq_wg + (size_t)(t + 1) * Dn,
                    (unsigned int)(uintptr_t)&yraw[(t + 1) & 1][0]);

    // ---- phase A: contract over w (this wave: x = j) ----
    float av[8];
#pragma unroll
    for (int r = 0; r < 8; ++r)
      av[r] = alpha_s[lp * 256 + (8 * hs + r) * 16 + j];
    float pm = av[0];
#pragma unroll
    for (int r = 1; r < 8; ++r) pm = fmaxf(pm, av[r]);
    const float m1 = fmaxf(pm, __shfl_xor(pm, 16, 32));
    v16h A1;
#pragma unroll
    for (int r = 0; r < 8; ++r) {
      const float ea = __expf(av[r] - m1);
      const _Float16 h = (_Float16)ea;
      A1[r] = h; A1[r + 8] = (_Float16)(ea - (float)h);
    }
    v8f C1 = {};
    C1 = wmma_f16(A1, pwB1, C1);
    C1 = wmma_f16(A1, pwB2, C1);
#pragma unroll
    for (int r = 0; r < 8; ++r) {
      const int b = r + 8 * hs;
      const float mm = __shfl(m1, b, 32);
      aw_s[b * 256 + lp * 16 + j] = mm + __logf(C1[r]);
    }
    __syncthreads();

    // ---- phase B: contract over x (this wave: w' = j), + emission, update --
    float a2[8];
#pragma unroll
    for (int r = 0; r < 8; ++r)
      a2[r] = aw_s[lp * 256 + j * 16 + 8 * hs + r];
    float pm2 = a2[0];
#pragma unroll
    for (int r = 1; r < 8; ++r) pm2 = fmaxf(pm2, a2[r]);
    const float m2 = fmaxf(pm2, __shfl_xor(pm2, 16, 32));
    v16h A2;
#pragma unroll
    for (int r = 0; r < 8; ++r) {
      const float ea = __expf(a2[r] - m2);
      const _Float16 h = (_Float16)ea;
      A2[r] = h; A2[r + 8] = (_Float16)(ea - (float)h);
    }
    v8f C2 = {};
    C2 = wmma_f16(A2, pxB1, C2);
    C2 = wmma_f16(A2, pxB2, C2);

    // emission for timestep t from the DMA'd tile
    {
      v16h Y0, Y1;
      build_yA(&yraw[t & 1][lp * Dn], hs, Y0, Y1);
      v8f E = {};
      E = wmma_f16(Y0, dBh0, E);
      E = wmma_f16(Y1, dBh1, E);
      E = wmma_f16(Y0, dBl0, E);
      E = wmma_f16(Y1, dBl1, E);
#pragma unroll
      for (int r = 0; r < 8; ++r) {
        const int b = r + 8 * hs;
        const float mm = __shfl(m2, b, 32);
        const float an = mm + __logf(C2[r]) + E[r] + base_j;
        const int ai = b * 256 + j * 16 + lp;
        const float old = alpha_s[ai];
        alpha_s[ai] = (t < lenv[r]) ? an : old;  // masked steps carry alpha
      }
    }

    if ((t + 1 < Tn) && (j == 0))
      __builtin_amdgcn_s_wait_tensorcnt(0);  // DMA done before next iteration
    __syncthreads();
  }

  // ---------------- final: out[b] = logsumexp over 256 states ----------------
  {
    float vals[8];
    float mv = -INFINITY;
#pragma unroll
    for (int i = 0; i < 8; ++i) {
      vals[i] = alpha_s[j * 256 + lane + 32 * i];
      mv = fmaxf(mv, vals[i]);
    }
#pragma unroll
    for (int off = 16; off >= 1; off >>= 1)
      mv = fmaxf(mv, __shfl_xor(mv, off, 32));
    float s = 0.f;
#pragma unroll
    for (int i = 0; i < 8; ++i) s += __expf(vals[i] - mv);
#pragma unroll
    for (int off = 16; off >= 1; off >>= 1) s += __shfl_xor(s, off, 32);
    if (lane == 0) out[bg0 + j] = mv + __logf(s);
  }
}

extern "C" void kernel_launch(void* const* d_in, const int* in_sizes, int n_in,
                              void* d_out, int out_size, void* d_ws, size_t ws_size,
                              hipStream_t stream) {
  const float* seq     = (const float*)d_in[0];
  const int*   lengths = (const int*)d_in[1];
  const float* pw      = (const float*)d_in[2];
  const float* px      = (const float*)d_in[3];
  const float* py      = (const float*)d_in[4];
  float* out = (float*)d_out;
  fhmm_forward<<<Bn / BT, NTHREADS, 0, stream>>>(seq, lengths, pw, px, py, out);
}
